// DataONEEncoder_27504970563690
// MI455X (gfx1250) — matrix operand order repaired and checked by reading, not compiled, not run on hardware
//
#include <hip/hip_runtime.h>

// ---------------------------------------------------------------------------
// MI455X / gfx1250 implementation of the 2-layer GRU encoder.
//
// - All GEMMs: v_wmma_f32_16x16x32_f16 (f16 in, f32 accumulate).
// - Recurrences: 16 workgroups x 16 batch rows (WMMA M-tile), 32 waves each
//   owning a 16-wide H slice; h state ping-pongs in LDS (f16, A-fragment
//   friendly layout), h_old kept in registers per-lane.
// - Per-step gate slab gx[b0..b0+15][t][:] (96 KB) is double-buffered in LDS
//   and prefetched one step ahead with GLOBAL_LOAD_ASYNC_TO_LDS_B128 (inline
//   asm), waited with s_wait_asynccnt (separate ASYNCcnt counter -> overlaps
//   the WMMA chain without disturbing loadcnt waits).
// - Layer-1 recurrence fuses LayerNorm + (last + mean-over-T) pooling.
// ---------------------------------------------------------------------------

typedef __attribute__((ext_vector_type(16))) _Float16 f16x16;
typedef __attribute__((ext_vector_type(8)))  _Float16 f16x8;
typedef __attribute__((ext_vector_type(8)))  float    f32x8;

#define B_   256
#define T_   512
#define F_   65
#define H_   512
#define H3_  1536
#define KP0_ 160          // 2F=130 padded to a multiple of 32
#define HP_  520          // LDS h row stride (elems): 1040B, 16B-aligned, 4-bank skew
#define LN_EPS_ 1e-5f

#define LDS_AS __attribute__((address_space(3)))

// --- CDNA5 async copy (global -> LDS), tracked by ASYNCcnt ------------------
// VFLAT async form: VDST = 32-bit LDS byte address, VADDR = 64-bit global
// address, "off" = no SADDR (CDNA5 ISA, opcode 98).

static __device__ __forceinline__ void async_cp_b128(const float* g, float* l) {
  unsigned la = (unsigned)(unsigned long long)(LDS_AS void*)l;
  asm volatile("global_load_async_to_lds_b128 %0, %1, off"
               :: "v"(la), "v"(g) : "memory");
}

static __device__ __forceinline__ void wait_async_le(int n) {
#if __has_builtin(__builtin_amdgcn_s_wait_asynccnt)
  if (n == 0) __builtin_amdgcn_s_wait_asynccnt(0);
  else        __builtin_amdgcn_s_wait_asynccnt(6);
#else
  if (n == 0) asm volatile("s_wait_asynccnt 0" ::: "memory");
  else        asm volatile("s_wait_asynccnt 6" ::: "memory");
#endif
}

// --- WMMA fragment loaders (layouts per CDNA5 ISA §7.12.2) -------

// A: 16x32 f16 tile from a row-major matrix. lane&15 -> M row; lane>>4 picks
// the K=8..15 / 24..31 halves.
static __device__ __forceinline__ f16x16
load_afrag(const _Float16* base, int lane, int strideA, int k0) {
  const _Float16* p = base + (long)(lane & 15) * strideA + k0 + (lane >> 4) * 8;
  f16x16 a;
  *(f16x8*)(&a)       = *(const f16x8*)(p);        // VGPR0-3
  *(((f16x8*)&a) + 1) = *(const f16x8*)(p + 16);   // VGPR4-7
  return a;
}

// B: 32x16 f16 tile where B[k][n] = W[n0+n][k] (W row-major).
static __device__ __forceinline__ f16x16
load_bfrag(const _Float16* W, int n0, int strideB, int k0, int lane) {
  const _Float16* p = W + (long)(n0 + (lane & 15)) * strideB + k0 + (lane >> 4) * 8;
  f16x16 b;
  *(f16x8*)(&b)       = *(const f16x8*)(p);
  *(((f16x8*)&b) + 1) = *(const f16x8*)(p + 16);
  return b;
}

static __device__ __forceinline__ f32x8 wmma_f16(f16x16 a, f16x16 b, f32x8 c) {
  return __builtin_amdgcn_wmma_f32_16x16x32_f16(false, a, false, b, (short)0, c,
                                                false, false);
}

// --- packing / conversion kernels -------------------------------------------

__global__ void pack_xm(const float* __restrict__ x, const int* __restrict__ mask,
                        _Float16* __restrict__ out) {
  const long total = (long)B_ * T_ * KP0_;
  long i = (long)blockIdx.x * blockDim.x + threadIdx.x;
  if (i >= total) return;
  long m = i / KP0_;
  int  k = (int)(i - m * KP0_);
  float v;
  if (k < F_)          v = x[m * F_ + k];
  else if (k < 2 * F_) v = (float)mask[m * F_ + (k - F_)];
  else                 v = 0.f;
  out[i] = (_Float16)v;
}

__global__ void cvt_pad(const float* __restrict__ in, _Float16* __restrict__ out,
                        int rows, int K, int Kpad) {
  long total = (long)rows * Kpad;
  long i = (long)blockIdx.x * blockDim.x + threadIdx.x;
  if (i >= total) return;
  int r = (int)(i / Kpad);
  int k = (int)(i - (long)r * Kpad);
  out[i] = (_Float16)(k < K ? in[(long)r * K + k] : 0.f);
}

// --- input-projection GEMM: gx[m][n] = A[m][:].W[n][:] + bias[n] ------------
// 2x2 register blocking: each wave computes a 32x32 tile (4 WMMAs per 8 b128
// loads). grid (M/32, 6), 256 threads = 8 waves -> 48 N-pairs of 32.
__global__ void __launch_bounds__(256)
gemm_gates(const _Float16* __restrict__ A, const _Float16* __restrict__ W,
           const float* __restrict__ bias, float* __restrict__ gxout,
           int Kpad, int kfrags) {
  const int tid = threadIdx.x, lane = tid & 31, wave = tid >> 5;
  const int lmod = lane & 15, lhalf = lane >> 4;
  const long m0 = (long)blockIdx.x * 32;
  const int n0 = (blockIdx.y * 8 + wave) * 32;
  const _Float16* Ab0 = A + m0 * Kpad;
  const _Float16* Ab1 = A + (m0 + 16) * Kpad;
  f32x8 acc00 = {}, acc01 = {}, acc10 = {}, acc11 = {};
  for (int kf = 0; kf < kfrags; ++kf) {
    const int k0 = kf * 32;
    f16x16 a0 = load_afrag(Ab0, lane, Kpad, k0);
    f16x16 a1 = load_afrag(Ab1, lane, Kpad, k0);
    f16x16 b0 = load_bfrag(W, n0, Kpad, k0, lane);
    f16x16 b1 = load_bfrag(W, n0 + 16, Kpad, k0, lane);
    acc00 = wmma_f16(a0, b0, acc00);
    acc01 = wmma_f16(a0, b1, acc01);
    acc10 = wmma_f16(a1, b0, acc10);
    acc11 = wmma_f16(a1, b1, acc11);
  }
  const float bn0 = bias[n0 + lmod];
  const float bn1 = bias[n0 + 16 + lmod];
#pragma unroll
  for (int e = 0; e < 8; ++e) {
    const long mA = m0 + e + lhalf * 8;
    const long mB = mA + 16;
    gxout[mA * H3_ + n0 + lmod]      = acc00[e] + bn0;
    gxout[mA * H3_ + n0 + 16 + lmod] = acc01[e] + bn1;
    gxout[mB * H3_ + n0 + lmod]      = acc10[e] + bn0;
    gxout[mB * H3_ + n0 + 16 + lmod] = acc11[e] + bn1;
  }
}

// --- recurrent GRU kernel ----------------------------------------------------
template <int LAYER>
__global__ void __launch_bounds__(1024)
gru_rec(const float* __restrict__ gx, const _Float16* __restrict__ Whh,
        const float* __restrict__ bhh, _Float16* __restrict__ h_out,
        const float* __restrict__ ln_g, const float* __restrict__ ln_b,
        float* __restrict__ pooled_out) {
  extern __shared__ char smem[];
  _Float16* hF    = (_Float16*)smem;                           // [2][16][HP_] f16
  float*    gxL   = (float*)(smem + 2 * 16 * HP_ * 2);         // [2][16][1536] f32
  float*    hS    = gxL + 2 * 16 * H3_;                        // [16][HP_] (LAYER 1)
  float*    pool  = hS + 16 * HP_;                             // [16][512]
  float*    red_s = pool + 16 * 512;                           // [16][64]
  float*    red_q = red_s + 16 * 64;                           // [16][64]
  float*    mu_s  = red_q + 16 * 64;                           // [16]
  float*    rs_s  = mu_s + 16;                                 // [16]

  const int tid = threadIdx.x;
  const int lane = tid & 31, wave = tid >> 5;
  const int lmod = lane & 15, lhalf = lane >> 4;
  const int b0 = blockIdx.x * 16;
  const int j0 = wave * 16;          // this wave's H-slice start
  const int jc = j0 + lmod;          // this lane's H column

  // async-prefetch mapping: thread -> (row = tid>>6, 6 x 16B chunks of 1536 f32)
  const int prow = tid >> 6;
  const int pu0  = tid & 63;
  const float* gsrc_row = gx + (long)(b0 + prow) * T_ * H3_;
  float* lrow0 = gxL + prow * H3_;
  float* lrow1 = gxL + 16 * H3_ + prow * H3_;

  for (int i = tid; i < 2 * 16 * HP_; i += 1024) hF[i] = (_Float16)0.f;
  if (LAYER == 1)
    for (int i = tid; i < 16 * 512; i += 1024) pool[i] = 0.f;
  __syncthreads();

  const float bh_r = bhh[0 * H_ + jc];
  const float bh_z = bhh[1 * H_ + jc];
  const float bh_n = bhh[2 * H_ + jc];
  const float invT = 1.0f / (float)T_;

  float hold[8];
#pragma unroll
  for (int e = 0; e < 8; ++e) hold[e] = 0.f;

  // prologue: prefetch gx slab for t = 0 into buffer 0
  {
    const float* g = gsrc_row;
#pragma unroll
    for (int k = 0; k < 6; ++k) {
      int u = (pu0 + k * 64) * 4;
      async_cp_b128(g + u, lrow0 + u);
    }
  }

  for (int t = 0; t < T_; ++t) {
    const int p = t & 1;
    const _Float16* hFr = hF + p * 16 * HP_;
    _Float16*       hFw = hF + (1 - p) * 16 * HP_;
    float* gxcur  = gxL + p * 16 * H3_;
    float* lnext  = (p == 0) ? lrow1 : lrow0;

    // prefetch t+1's gx slab while the WMMA chain runs
    if (t + 1 < T_) {
      const float* g = gsrc_row + (long)(t + 1) * H3_;
#pragma unroll
      for (int k = 0; k < 6; ++k) {
        int u = (pu0 + k * 64) * 4;
        async_cp_b128(g + u, lnext + u);
      }
    }

    // gh tiles for the three gates of this wave's H-slice
    f32x8 acc_r = {}, acc_z = {}, acc_n = {};
#pragma unroll 4
    for (int kf = 0; kf < H_ / 32; ++kf) {
      f16x16 a  = load_afrag(hFr, lane, HP_, kf * 32);
      f16x16 br = load_bfrag(Whh, 0 * H_ + j0, H_, kf * 32, lane);
      f16x16 bz = load_bfrag(Whh, 1 * H_ + j0, H_, kf * 32, lane);
      f16x16 bn = load_bfrag(Whh, 2 * H_ + j0, H_, kf * 32, lane);
      acc_r = wmma_f16(a, br, acc_r);
      acc_z = wmma_f16(a, bz, acc_z);
      acc_n = wmma_f16(a, bn, acc_n);
    }

    // this step's gx slab is ready once our own async ops retire + barrier
    wait_async_le((t + 1 < T_) ? 6 : 0);
    __syncthreads();

    // elementwise gate math + state update (h_old lives in registers)
#pragma unroll
    for (int e = 0; e < 8; ++e) {
      const int m = e + lhalf * 8;
      float i_r = gxcur[m * H3_ + 0 * H_ + jc];
      float i_z = gxcur[m * H3_ + 1 * H_ + jc];
      float i_n = gxcur[m * H3_ + 2 * H_ + jc];
      float r = 1.f / (1.f + __expf(-(i_r + acc_r[e] + bh_r)));
      float z = 1.f / (1.f + __expf(-(i_z + acc_z[e] + bh_z)));
      float nn = tanhf(i_n + r * (acc_n[e] + bh_n));
      float hnew = (1.f - z) * nn + z * hold[e];
      hold[e] = hnew;
      hFw[m * HP_ + jc] = (_Float16)hnew;
      if (LAYER == 1) hS[m * HP_ + jc] = hnew;
      if (LAYER == 0)
        h_out[((long)(b0 + m) * T_ + t) * H_ + jc] = (_Float16)hnew;
    }
    __syncthreads();

    if (LAYER == 1) {
      // fused LayerNorm + pooling accumulation over the fresh h rows
      const int row = tid >> 6;
      const int c64 = tid & 63;
      float s = 0.f, q = 0.f;
      for (int jj = c64; jj < H_; jj += 64) {
        float v = hS[row * HP_ + jj];
        s += v;
        q += v * v;
      }
      red_s[row * 64 + c64] = s;
      red_q[row * 64 + c64] = q;
      __syncthreads();
      if (tid < 16) {
        float ss = 0.f, qq = 0.f;
        for (int i = 0; i < 64; ++i) {
          ss += red_s[tid * 64 + i];
          qq += red_q[tid * 64 + i];
        }
        float mu = ss * (1.f / (float)H_);
        float var = qq * (1.f / (float)H_) - mu * mu;
        mu_s[tid] = mu;
        rs_s[tid] = rsqrtf(var + LN_EPS_);
      }
      __syncthreads();
      const float wgt = invT + ((t == T_ - 1) ? 1.f : 0.f);
      const float mu = mu_s[row], rs = rs_s[row];
      for (int jj = c64; jj < H_; jj += 64) {
        float v = (hS[row * HP_ + jj] - mu) * rs * ln_g[jj] + ln_b[jj];
        pool[row * 512 + jj] += wgt * v;
      }
      __syncthreads();
    }
  }

  if (LAYER == 1) {
    for (int i = tid; i < 16 * 512; i += 1024) {
      int row = i >> 9, jj = i & 511;
      pooled_out[(long)(b0 + row) * H_ + jj] = pool[i];
    }
  }
}

// --- final projection + exact-erf GELU --------------------------------------
__global__ void __launch_bounds__(256)
proj_gelu(const _Float16* __restrict__ A, const _Float16* __restrict__ Wp,
          const float* __restrict__ bp, float* __restrict__ out) {
  const int tid = threadIdx.x, lane = tid & 31, wave = tid >> 5;
  const int lmod = lane & 15, lhalf = lane >> 4;
  const int m0 = blockIdx.x * 16;
  const int n0 = (blockIdx.y * 8 + wave) * 16;
  const _Float16* Ab = A + (long)m0 * H_;
  f32x8 acc = {};
  for (int kf = 0; kf < H_ / 32; ++kf) {
    f16x16 a = load_afrag(Ab, lane, H_, kf * 32);
    f16x16 b = load_bfrag(Wp, n0, H_, kf * 32, lane);
    acc = wmma_f16(a, b, acc);
  }
  const float bn = bp[n0 + lmod];
#pragma unroll
  for (int e = 0; e < 8; ++e) {
    const int m = m0 + e + lhalf * 8;
    float y = acc[e] + bn;
    float g = 0.5f * y * (1.f + erff(y * 0.70710678118654752f));
    out[m * (H_ / 2) + n0 + lmod] = g;
  }
}

// ---------------------------------------------------------------------------

extern "C" void kernel_launch(void* const* d_in, const int* in_sizes, int n_in,
                              void* d_out, int out_size, void* d_ws, size_t ws_size,
                              hipStream_t stream) {
  (void)in_sizes; (void)n_in; (void)out_size; (void)ws_size;
  const float* x      = (const float*)d_in[0];
  const int*   mask   = (const int*)d_in[1];
  const float* W_ih0  = (const float*)d_in[2];
  const float* W_hh0  = (const float*)d_in[3];
  const float* b_ih0  = (const float*)d_in[4];
  const float* b_hh0  = (const float*)d_in[5];
  const float* W_ih1  = (const float*)d_in[6];
  const float* W_hh1  = (const float*)d_in[7];
  const float* b_ih1  = (const float*)d_in[8];
  const float* b_hh1  = (const float*)d_in[9];
  const float* ln_g   = (const float*)d_in[10];
  const float* ln_b   = (const float*)d_in[11];
  const float* W_proj = (const float*)d_in[12];
  const float* b_proj = (const float*)d_in[13];
  float* out = (float*)d_out;

  const long M = (long)B_ * T_;   // 131072
  char* ws = (char*)d_ws;
  size_t off = 0;
  auto alloc = [&](size_t bytes) -> void* {
    void* p = ws + off;
    off += (bytes + 255) & ~(size_t)255;
    return p;
  };
  _Float16* xm    = (_Float16*)alloc(M * KP0_ * 2);
  _Float16* wi0   = (_Float16*)alloc((size_t)H3_ * KP0_ * 2);
  _Float16* wh0   = (_Float16*)alloc((size_t)H3_ * H_ * 2);
  _Float16* wi1   = (_Float16*)alloc((size_t)H3_ * H_ * 2);
  _Float16* wh1   = (_Float16*)alloc((size_t)H3_ * H_ * 2);
  _Float16* wpr   = (_Float16*)alloc((size_t)(H_ / 2) * H_ * 2);
  float*    gx    = (float*)alloc((size_t)M * H3_ * 4);       // reused L0+L1
  _Float16* h1    = (_Float16*)alloc((size_t)M * H_ * 2);
  float*    pool  = (float*)alloc((size_t)B_ * H_ * 4);
  _Float16* poolh = (_Float16*)alloc((size_t)B_ * H_ * 2);

  // 1) pack x||mask (f16, K 130->160) and convert weights to f16
  {
    long tot = M * KP0_;
    pack_xm<<<dim3((unsigned)((tot + 255) / 256)), 256, 0, stream>>>(x, mask, xm);
  }
  {
    long tot = (long)H3_ * KP0_;
    cvt_pad<<<dim3((unsigned)((tot + 255) / 256)), 256, 0, stream>>>(W_ih0, wi0, H3_, 2 * F_, KP0_);
  }
  {
    long tot = (long)H3_ * H_;
    cvt_pad<<<dim3((unsigned)((tot + 255) / 256)), 256, 0, stream>>>(W_hh0, wh0, H3_, H_, H_);
    cvt_pad<<<dim3((unsigned)((tot + 255) / 256)), 256, 0, stream>>>(W_ih1, wi1, H3_, H_, H_);
    cvt_pad<<<dim3((unsigned)((tot + 255) / 256)), 256, 0, stream>>>(W_hh1, wh1, H3_, H_, H_);
  }
  {
    long tot = (long)(H_ / 2) * H_;
    cvt_pad<<<dim3((unsigned)((tot + 255) / 256)), 256, 0, stream>>>(W_proj, wpr, H_ / 2, H_, H_);
  }

  const size_t lds0 = (size_t)2 * 16 * HP_ * 2            // hF ping-pong
                    + (size_t)2 * 16 * H3_ * 4;           // gx ping-pong  = 229,888 B
  const size_t lds1 = lds0
                    + (size_t)16 * HP_ * 4                // hS
                    + (size_t)16 * 512 * 4                // pooled
                    + (size_t)2 * 16 * 64 * 4             // reduction
                    + (size_t)2 * 16 * 4;                 // mu/rs = 304,256 B (<320KB)

  // 2) layer 0: input GEMM then recurrence (emits h1 in f16)
  gemm_gates<<<dim3((unsigned)(M / 32), 6), 256, 0, stream>>>(xm, wi0, b_ih0, gx, KP0_, KP0_ / 32);
  gru_rec<0><<<dim3(B_ / 16), 1024, lds0, stream>>>(gx, wh0, b_hh0, h1, nullptr, nullptr, nullptr);

  // 3) layer 1: input GEMM (gx reused) then recurrence fused with LN+pooling
  gemm_gates<<<dim3((unsigned)(M / 32), 6), 256, 0, stream>>>(h1, wi1, b_ih1, gx, H_, H_ / 32);
  gru_rec<1><<<dim3(B_ / 16), 1024, lds1, stream>>>(gx, wh1, b_hh1, nullptr, ln_g, ln_b, pool);

  // 4) projection + exact GELU
  {
    long tot = (long)B_ * H_;
    cvt_pad<<<dim3((unsigned)((tot + 255) / 256)), 256, 0, stream>>>(pool, poolh, B_, H_, H_);
  }
  proj_gelu<<<dim3(B_ / 16, 2), 256, 0, stream>>>(poolh, wpr, b_proj, out);
}
